// Attention_64665027609055
// MI455X (gfx1250) — matrix operand order repaired and checked by reading, not compiled, not run on hardware
//
#include <hip/hip_runtime.h>

// Problem constants (from reference): B=2, S=2048, D=1024, H=16, head_dim=64
#define BB 2
#define SS 2048
#define DD 1024
#define HH 16
#define HD 64
#define MM (BB * SS) // 4096 rows total

typedef __bf16 bf16;
typedef __attribute__((ext_vector_type(16))) __bf16 v16bf;
typedef __attribute__((ext_vector_type(8)))  __bf16 v8bf;
typedef __attribute__((ext_vector_type(4)))  __bf16 v4bf;
typedef __attribute__((ext_vector_type(8)))  float  v8f;

__device__ __forceinline__ v8f wmma_bf16(v16bf a, v16bf b, v8f c) {
  // D = A(16x32 bf16) * B(32x16 bf16) + C(16x16 f32)
  return __builtin_amdgcn_wmma_f32_16x16x32_bf16(false, a, false, b,
                                                 (short)0, c, false, false);
}

__device__ __forceinline__ v8f zero8() {
  v8f z;
#pragma unroll
  for (int i = 0; i < 8; ++i) z[i] = 0.0f;
  return z;
}

__device__ __forceinline__ v16bf cat8(v8bf lo, v8bf hi) {
  return __builtin_shufflevector(lo, hi, 0, 1, 2, 3, 4, 5, 6, 7,
                                 8, 9, 10, 11, 12, 13, 14, 15);
}

// A fragment (16x32 bf16) from row-major [row][k] storage (stride mult. of 8).
// Lane L(0-15): row=L, K {0..7,16..23}; lane L+16: row=L, K {8..15,24..31}.
// Two contiguous 16B runs per lane -> 2x ds/global b128.
__device__ __forceinline__ v16bf frag_a(const bf16* p0, int stride, int lane) {
  const bf16* p = p0 + (lane & 15) * stride + ((lane >> 4) << 3);
  return cat8(*(const v8bf*)(p), *(const v8bf*)(p + 16));
}

// B fragment (32x16 bf16): lane L(0-15): col n=L, K {0..7,16..23}; lane L+16:
// col n=L, K {8..15,24..31}; source [k][n] (strided gather).
__device__ __forceinline__ v16bf frag_b_kn(const bf16* p0, int stride, int lane) {
  const bf16* p = p0 + (lane & 15) + ((lane >> 4) << 3) * stride;
  v16bf f;
#pragma unroll
  for (int s = 0; s < 16; ++s) f[s] = p[(s < 8 ? s : s + 8) * stride];
  return f;
}

// ---------------------------------------------------------------- converters
__global__ __launch_bounds__(256)
void cvt_f32_to_bf16_4(const float4* __restrict__ src, bf16* __restrict__ dst,
                       int n4) {
  int i = blockIdx.x * 256 + threadIdx.x;
  if (i < n4) {
    float4 v = src[i];
    v4bf o;
    o[0] = (bf16)v.x; o[1] = (bf16)v.y; o[2] = (bf16)v.z; o[3] = (bf16)v.w;
    *(v4bf*)(dst + 4 * (size_t)i) = o;
  }
}

// Wt[d][o] = (bf16)W[o][d]   (D x D), 32x32 LDS tiles, 256 threads.
__global__ __launch_bounds__(256)
void wT_bf16(const float* __restrict__ W, bf16* __restrict__ Wt) {
  __shared__ float tile[32][33];
  const int bx = blockIdx.x * 32, by = blockIdx.y * 32;
  const int tx = threadIdx.x & 31, ty = threadIdx.x >> 5; // 32 x 8
#pragma unroll
  for (int i = 0; i < 32; i += 8)
    tile[ty + i][tx] = W[(size_t)(by + ty + i) * DD + bx + tx];
  __syncthreads();
#pragma unroll
  for (int i = 0; i < 32; i += 8)
    Wt[(size_t)(bx + ty + i) * DD + by + tx] = (bf16)tile[tx][ty + i];
}

// ------------------------------------------------------------------- GEMM
// C = A(M x D) * Wt(D x D, k-major) + bias.
// MODE 0: write bf16 head-split [B,H,S,64]; MODE 1: write fp32 [M,D].
// 128x128 block tile, BK=32, 256 threads = 8 waves (4M x 2N), wave = 32x64.
template <int MODE>
__global__ __launch_bounds__(256)
void gemm128(const bf16* __restrict__ A, const bf16* __restrict__ Bt,
             const float* __restrict__ bias, void* __restrict__ Cout) {
  __shared__ __align__(16) bf16 sA[128][40];
  __shared__ __align__(16) bf16 sB[32][136];

  const int m0 = blockIdx.x * 128;
  const int n0 = blockIdx.y * 128;
  const int t = threadIdx.x;
  const int lane = t & 31, wave = t >> 5;
  const int wm = (wave & 3) * 32, wn = (wave >> 2) * 64;

  v8f acc[2][4];
#pragma unroll
  for (int i = 0; i < 2; ++i)
#pragma unroll
    for (int j = 0; j < 4; ++j) acc[i][j] = zero8();

  const int ar = t >> 1, ac = (t & 1) * 16;  // A tile: 128x32
  const int br = t >> 3, bc = (t & 7) * 16;  // B tile: 32x128

  for (int k0 = 0; k0 < DD; k0 += 32) {
    const bf16* ap = A + (size_t)(m0 + ar) * DD + k0 + ac;
    *(v8bf*)&sA[ar][ac]     = *(const v8bf*)(ap);
    *(v8bf*)&sA[ar][ac + 8] = *(const v8bf*)(ap + 8);
    const bf16* bp = Bt + (size_t)(k0 + br) * DD + n0 + bc;
    *(v8bf*)&sB[br][bc]     = *(const v8bf*)(bp);
    *(v8bf*)&sB[br][bc + 8] = *(const v8bf*)(bp + 8);
    if (k0 + 32 < DD) {               // pull next k-tile toward the WGP
      __builtin_prefetch(ap + 32, 0, 1);
      __builtin_prefetch(bp + (size_t)32 * DD, 0, 1);
    }
    __syncthreads();

    v16bf afr[2], bfr[4];
#pragma unroll
    for (int i = 0; i < 2; ++i) afr[i] = frag_a(&sA[wm + i * 16][0], 40, lane);
#pragma unroll
    for (int j = 0; j < 4; ++j)
      bfr[j] = frag_b_kn(&sB[0][wn + j * 16], 136, lane);
#pragma unroll
    for (int i = 0; i < 2; ++i)
#pragma unroll
      for (int j = 0; j < 4; ++j)
        acc[i][j] = wmma_bf16(afr[i], bfr[j], acc[i][j]);
    __syncthreads();
  }

  const int cl = lane & 15, half = lane >> 4;
#pragma unroll
  for (int i = 0; i < 2; ++i) {
#pragma unroll
    for (int j = 0; j < 4; ++j) {
      const int col = n0 + wn + j * 16 + cl;
      const float bv = bias[col];
#pragma unroll
      for (int r = 0; r < 8; ++r) {
        const int row = m0 + wm + i * 16 + r + 8 * half;
        const float v = acc[i][j][r] + bv;
        if (MODE == 0) {
          const int b = row >> 11, s = row & (SS - 1);
          const int h = col >> 6, e = col & (HD - 1);
          ((bf16*)Cout)[(((size_t)(b * HH + h) * SS + s) << 6) + e] = (bf16)v;
        } else {
          ((float*)Cout)[(size_t)row * DD + col] = v;
        }
      }
    }
  }
}

// --------------------------------------------------------- flash attention
// Block = (b, h, 64 q rows); 128 threads = 4 waves; wave owns 16 q rows.
// Score tile computed TRANSPOSED: S^T = K * Q^T, so the per-tile A fragment
// (K) is a contiguous vectorized LDS read and the strided Q^T B-fragment is
// loaded once per kernel. Mask is pre-expanded once per block into an additive
// float bias in LDS (masked -> -9e9), so masking is a single FMA per score.
// Per-lane online softmax (lane owns q column = lane&15, kpos halves combined
// with one shfl_xor(16)). P^T restaged q-major with 4x ds_store_b128 per lane,
// then standard-orientation O += P*V.
__global__ __launch_bounds__(128)
void attn64(const bf16* __restrict__ Qg, const bf16* __restrict__ Kg,
            const bf16* __restrict__ Vg, const unsigned char* __restrict__ mask,
            bf16* __restrict__ AO) {
  __shared__ __align__(16) bf16 sK[64][72];
  __shared__ __align__(16) bf16 sV[64][72];
  __shared__ __align__(16) bf16 sP[4][16][72];
  __shared__ __align__(16) float sBias[SS]; // additive key-mask bias, 8 KB

  const int b = blockIdx.z, h = blockIdx.y, q0 = blockIdx.x * 64;
  const int t = threadIdx.x, lane = t & 31, w = t >> 5;
  const int cl = lane & 15, half = lane >> 4;
  const size_t bh = (size_t)(b * HH + h) * SS;

  // expand bool mask -> float bias once (16 bytes -> 16 floats per thread)
  {
    const uchar4* mp = (const uchar4*)(mask + b * SS + t * 16);
#pragma unroll
    for (int c = 0; c < 4; ++c) {
      const uchar4 m4 = mp[c];
      float4 f;
      f.x = m4.x ? -9.0e9f : 0.0f;
      f.y = m4.y ? -9.0e9f : 0.0f;
      f.z = m4.z ? -9.0e9f : 0.0f;
      f.w = m4.w ? -9.0e9f : 0.0f;
      ((float4*)&sBias[t * 16])[c] = f;
    }
  }

  // Q^T B-fragments (lane&15 = q row, slots = hd in table order within 32-chunk).
  // Strided gather, but only once per kernel (amortized over 32 K-tiles).
  v16bf qbf[2];
#pragma unroll
  for (int kk = 0; kk < 2; ++kk) {
    const bf16* qp = Qg + ((bh + q0 + w * 16 + cl) << 6) + kk * 32 + 8 * half;
    v16bf f;
#pragma unroll
    for (int n = 0; n < 16; ++n) f[n] = qp[n < 8 ? n : n + 8];
    qbf[kk] = f;
  }

  v8f oacc[4]; // O: rows = q (16), cols = hd sub-tile jj*16
#pragma unroll
  for (int j = 0; j < 4; ++j) oacc[j] = zero8();
  float m = -3.0e38f, l = 0.0f; // per-lane stats for q = cl

  const float scale = 0.125f; // 1/sqrt(64)
  const int lr = t >> 1, lc = (t & 1) * 32;

  for (int kt = 0; kt < SS; kt += 64) {
    __syncthreads(); // also covers the sBias fill on the first iteration
    {
      const bf16* kp = Kg + ((bh + kt + lr) << 6) + lc;
      const bf16* vp = Vg + ((bh + kt + lr) << 6) + lc;
#pragma unroll
      for (int c = 0; c < 32; c += 8) {
        *(v8bf*)&sK[lr][lc + c] = *(const v8bf*)(kp + c);
        *(v8bf*)&sV[lr][lc + c] = *(const v8bf*)(vp + c);
      }
      if (kt + 64 < SS) {
        __builtin_prefetch(kp + 64 * HD, 0, 1);
        __builtin_prefetch(vp + 64 * HD, 0, 1);
      }
    }
    __syncthreads();

    // S^T tile: sacc[jj] = K(sub-tile jj) * Q^T ; rows = kpos, cols = q
    v8f sacc[4];
#pragma unroll
    for (int jj = 0; jj < 4; ++jj) sacc[jj] = zero8();
#pragma unroll
    for (int kk = 0; kk < 2; ++kk)
#pragma unroll
      for (int jj = 0; jj < 4; ++jj) {
        v16bf kf = frag_a(&sK[jj * 16][kk * 32], 72, lane);
        sacc[jj] = wmma_bf16(kf, qbf[kk], sacc[jj]);
      }

    // masked scale: one FMA per score, bias streamed from LDS (b128 pairs)
    float pvv[4][8];
    float mloc = -3.0e38f;
#pragma unroll
    for (int jj = 0; jj < 4; ++jj) {
      const float* bp = &sBias[kt + jj * 16 + 8 * half];
#pragma unroll
      for (int r = 0; r < 8; ++r) {
        const float v = fmaf(sacc[jj][r], scale, bp[r]);
        pvv[jj][r] = v;
        mloc = fmaxf(mloc, v);
      }
    }
    mloc = fmaxf(mloc, __shfl_xor(mloc, 16, 32));
    const float mnew = fmaxf(m, mloc);
    const float fac = __expf(m - mnew);
    float rs = 0.0f;
#pragma unroll
    for (int jj = 0; jj < 4; ++jj)
#pragma unroll
      for (int r = 0; r < 8; ++r) {
        const float e = __expf(pvv[jj][r] - mnew);
        pvv[jj][r] = e;
        rs += e;
      }
    rs += __shfl_xor(rs, 16, 32);
    l = l * fac + rs;
    m = mnew;

    // stage P^T -> sP[w][q][kpos]: 8 contiguous kpos per jj = 1x b128 store
#pragma unroll
    for (int jj = 0; jj < 4; ++jj) {
      v8bf pk;
#pragma unroll
      for (int r = 0; r < 8; ++r) pk[r] = (bf16)pvv[jj][r];
      *(v8bf*)&sP[w][cl][jj * 16 + 8 * half] = pk;
    }

    // rescale O: row q = r + 8*half; its factor lives in lane q
    float f8[8];
#pragma unroll
    for (int r = 0; r < 8; ++r) f8[r] = __shfl(fac, r + 8 * half, 32);
#pragma unroll
    for (int jj = 0; jj < 4; ++jj)
#pragma unroll
      for (int r = 0; r < 8; ++r) oacc[jj][r] *= f8[r];
    __syncthreads();

    // O += P * V (A = P from q-major sP, B = V from [kpos][hd]; P via b128, V strided gather)
#pragma unroll
    for (int kk = 0; kk < 2; ++kk) {
      v16bf pf = frag_a(&sP[w][0][kk * 32], 72, lane);
#pragma unroll
      for (int jj = 0; jj < 4; ++jj) {
        v16bf vf = frag_b_kn(&sV[kk * 32][jj * 16], 72, lane);
        oacc[jj] = wmma_bf16(pf, vf, oacc[jj]);
      }
    }
  }

  // normalize and write merged-head bf16 [B,S,D] for the output projection
  const float linv = 1.0f / l;
  float i8[8];
#pragma unroll
  for (int r = 0; r < 8; ++r) i8[r] = __shfl(linv, r + 8 * half, 32);
#pragma unroll
  for (int r = 0; r < 8; ++r) {
    const int row = q0 + w * 16 + r + 8 * half;
#pragma unroll
    for (int jj = 0; jj < 4; ++jj)
      AO[((size_t)(b * SS + row) << 10) + h * 64 + jj * 16 + cl] =
          (bf16)(oacc[jj][r] * i8[r]);
  }
}

// ------------------------------------------------------------------ driver
extern "C" void kernel_launch(void* const* d_in, const int* in_sizes, int n_in,
                              void* d_out, int out_size, void* d_ws,
                              size_t ws_size, hipStream_t stream) {
  (void)in_sizes; (void)n_in; (void)out_size; (void)ws_size;
  const float* query = (const float*)d_in[0];
  const float* key   = (const float*)d_in[1];
  const float* value = (const float*)d_in[2];
  const unsigned char* mask = (const unsigned char*)d_in[3]; // bool array
  const float* Wq = (const float*)d_in[4];
  const float* bq = (const float*)d_in[5];
  const float* Wk = (const float*)d_in[6];
  const float* bk = (const float*)d_in[7];
  const float* Wv = (const float*)d_in[8];
  const float* bv = (const float*)d_in[9];
  const float* Wo = (const float*)d_in[10];
  const float* bo = (const float*)d_in[11];

  const size_t MD = (size_t)MM * DD; // 4M elems
  const size_t D2 = (size_t)DD * DD; // 1M elems
  bf16* p = (bf16*)d_ws;
  bf16* xq = p;  p += MD;
  bf16* xk = p;  p += MD;
  bf16* xv = p;  p += MD;
  bf16* wqt = p; p += D2;
  bf16* wkt = p; p += D2;
  bf16* wvt = p; p += D2;
  bf16* wot = p; p += D2;
  bf16* Qp = p;  p += MD;
  bf16* Kp = p;  p += MD;
  bf16* Vp = p;  p += MD;
  bf16* AO = xq; // x buffers are dead after the projections; reuse for attn out

  { // fp32 -> bf16 activation copies
    const int n4 = (int)(MD / 4);
    const int g = (n4 + 255) / 256;
    cvt_f32_to_bf16_4<<<g, 256, 0, stream>>>((const float4*)query, xq, n4);
    cvt_f32_to_bf16_4<<<g, 256, 0, stream>>>((const float4*)key,   xk, n4);
    cvt_f32_to_bf16_4<<<g, 256, 0, stream>>>((const float4*)value, xv, n4);
  }
  { // weight transpose + bf16
    dim3 g(DD / 32, DD / 32);
    wT_bf16<<<g, 256, 0, stream>>>(Wq, wqt);
    wT_bf16<<<g, 256, 0, stream>>>(Wk, wkt);
    wT_bf16<<<g, 256, 0, stream>>>(Wv, wvt);
    wT_bf16<<<g, 256, 0, stream>>>(Wo, wot);
  }
  { // Q/K/V projections -> head-split bf16
    dim3 g(MM / 128, DD / 128);
    gemm128<0><<<g, 256, 0, stream>>>(xq, wqt, bq, Qp);
    gemm128<0><<<g, 256, 0, stream>>>(xk, wkt, bk, Kp);
    gemm128<0><<<g, 256, 0, stream>>>(xv, wvt, bv, Vp);
  }
  { // fused masked-softmax attention
    dim3 g(SS / 64, HH, BB);
    attn64<<<g, 128, 0, stream>>>(Qp, Kp, Vp, mask, AO);
  }
  { // output projection -> fp32
    dim3 g(MM / 128, DD / 128);
    gemm128<1><<<g, 256, 0, stream>>>(AO, wot, bo, d_out);
  }
}
